// GAT_56023553409630
// MI455X (gfx1250) — matrix operand order repaired and checked by reading, compiled, not run on hardware
//
#include <hip/hip_runtime.h>
#include <math.h>

typedef __attribute__((ext_vector_type(16))) __bf16 v16bf;
typedef __attribute__((ext_vector_type(8)))  float  v8f;

#define NN 30000
#define NE 400000

// -------------------- generic fill --------------------
__global__ void k_fill(float* __restrict__ p, float v, int n) {
    int i = blockIdx.x * blockDim.x + threadIdx.x;
    if (i < n) p[i] = v;
}

// -------------------- weight pre-swizzle into WMMA B-fragment order --------------------
// Packed layout (per matrix t of K x N): frag(kt,nt) occupies 512 bf16:
//   Wp[((kt*(N/16) + nt)*32 + lane)*16 + e] = (bf16) W[k, n]
//   with k = kt*32 + 8*(lane>>4) + (e&7) + 16*(e>>3),  n = nt*16 + (lane&15)
__global__ void k_pack_w(const float* __restrict__ W, __bf16* __restrict__ Wp,
                         int K, int N, int T)
{
    int i = blockIdx.x * blockDim.x + threadIdx.x;
    if (i >= T * K * N) return;
    int t = i / (K * N);
    int r = i % (K * N);
    int e    = r & 15;
    int lane = (r >> 4) & 31;
    int tile = r >> 9;
    int NT = N / 16;
    int kt = tile / NT, nt = tile % NT;
    int k = kt * 32 + ((lane >> 4) * 8) + (e & 7) + ((e >> 3) * 16);
    int n = nt * 16 + (lane & 15);
    Wp[i] = (__bf16)W[(size_t)t * K * N + (size_t)k * N + n];
}

// -------------------- A fragment (row-major A[M,K], fp32 -> bf16) --------------------
// lane: m = mr, ks = {base..base+7, 16+base..16+base+7}, base = half*8
__device__ inline v16bf load_a_frag(const float* __restrict__ A, int ldA,
                                    int mr, int k0, int half, bool relu) {
    const float* ar = A + (size_t)mr * ldA + k0 + half * 8;
    float4 f0 = *(const float4*)(ar);
    float4 f1 = *(const float4*)(ar + 4);
    float4 f2 = *(const float4*)(ar + 16);
    float4 f3 = *(const float4*)(ar + 20);
    if (relu) {
        f0.x = fmaxf(f0.x, 0.f); f0.y = fmaxf(f0.y, 0.f); f0.z = fmaxf(f0.z, 0.f); f0.w = fmaxf(f0.w, 0.f);
        f1.x = fmaxf(f1.x, 0.f); f1.y = fmaxf(f1.y, 0.f); f1.z = fmaxf(f1.z, 0.f); f1.w = fmaxf(f1.w, 0.f);
        f2.x = fmaxf(f2.x, 0.f); f2.y = fmaxf(f2.y, 0.f); f2.z = fmaxf(f2.z, 0.f); f2.w = fmaxf(f2.w, 0.f);
        f3.x = fmaxf(f3.x, 0.f); f3.y = fmaxf(f3.y, 0.f); f3.z = fmaxf(f3.z, 0.f); f3.w = fmaxf(f3.w, 0.f);
    }
    v16bf a;
    a[0]=(__bf16)f0.x;  a[1]=(__bf16)f0.y;  a[2]=(__bf16)f0.z;  a[3]=(__bf16)f0.w;
    a[4]=(__bf16)f1.x;  a[5]=(__bf16)f1.y;  a[6]=(__bf16)f1.z;  a[7]=(__bf16)f1.w;
    a[8]=(__bf16)f2.x;  a[9]=(__bf16)f2.y;  a[10]=(__bf16)f2.z; a[11]=(__bf16)f2.w;
    a[12]=(__bf16)f3.x; a[13]=(__bf16)f3.y; a[14]=(__bf16)f3.z; a[15]=(__bf16)f3.w;
    return a;
}

// -------------------- projection GEMM: C[t] = A[t] @ W[t] + b[t] --------------------
// One wave computes a 16 x N strip; A fragment loaded once per k-step.
template<int K, int N>
__global__ __launch_bounds__(32) void k_proj_wmma(
    const float* __restrict__ A0, const float* __restrict__ A1,
    const float* __restrict__ A2, const float* __restrict__ A3,
    const __bf16* __restrict__ Wp, const float* __restrict__ bias,
    float* __restrict__ C)
{
    constexpr int NT = N / 16;
    const int t = blockIdx.y;
    const float* A = (t == 0) ? A0 : (t == 1) ? A1 : (t == 2) ? A2 : A3;
    const __bf16* Wt = Wp + (size_t)t * K * N;
    const float* bt = bias + (size_t)t * N;
    float* Ct = C + (size_t)t * NN * N;

    const int lane = threadIdx.x & 31;
    const int half = lane >> 4;
    const int m0 = blockIdx.x * 16;
    const int mr = m0 + (lane & 15);

    v8f acc[NT] = {};
    for (int k0 = 0; k0 < K; k0 += 32) {
        v16bf a = load_a_frag(A, K, mr, k0, half, false);
        if (k0 + 32 < K)
            __builtin_prefetch(A + (size_t)mr * K + k0 + 32 + half * 8, 0, 3);
        const int kt = k0 >> 5;
#pragma unroll
        for (int nt = 0; nt < NT; nt++) {
            v16bf b = *(const v16bf*)(Wt + (((size_t)kt * NT + nt) * 32 + lane) * 16);
            acc[nt] = __builtin_amdgcn_wmma_f32_16x16x32_bf16(false, a, false, b,
                                                              (short)0, acc[nt], false, false);
        }
    }
#pragma unroll
    for (int nt = 0; nt < NT; nt++) {
        const int nc = nt * 16 + (lane & 15);
        const float bb = bt[nc];
#pragma unroll
        for (int j = 0; j < 8; j++) {
            int m = m0 + j + half * 8;
            Ct[(size_t)m * N + nc] = acc[nt][j] + bb;
        }
    }
}

// ---- semantic score: score += sum_{n,c} tanh((relu(A)@kW+kb)[n,c]) * q[c] / NN ----
template<int K>   // kW is K x K
__global__ __launch_bounds__(32) void k_sem_score_wmma(
    const float* __restrict__ A, const __bf16* __restrict__ Wp,
    const float* __restrict__ kb, const float* __restrict__ q,
    float* __restrict__ score, float invN)
{
    constexpr int N = K, NT = N / 16;
    const int lane = threadIdx.x & 31;
    const int half = lane >> 4;
    const int m0 = blockIdx.x * 16;
    const int mr = m0 + (lane & 15);

    v8f acc[NT] = {};
    for (int k0 = 0; k0 < K; k0 += 32) {
        v16bf a = load_a_frag(A, K, mr, k0, half, true);   // relu fused on input
        if (k0 + 32 < K)
            __builtin_prefetch(A + (size_t)mr * K + k0 + 32 + half * 8, 0, 3);
        const int kt = k0 >> 5;
#pragma unroll
        for (int nt = 0; nt < NT; nt++) {
            v16bf b = *(const v16bf*)(Wp + (((size_t)kt * NT + nt) * 32 + lane) * 16);
            acc[nt] = __builtin_amdgcn_wmma_f32_16x16x32_bf16(false, a, false, b,
                                                              (short)0, acc[nt], false, false);
        }
    }
    float part = 0.f;
#pragma unroll
    for (int nt = 0; nt < NT; nt++) {
        const int nc = nt * 16 + (lane & 15);
        const float bn = kb[nc];
        const float qn = q[nc];
#pragma unroll
        for (int j = 0; j < 8; j++) part += tanhf(acc[nt][j] + bn) * qn;
    }
    part *= invN;
#pragma unroll
    for (int off = 16; off > 0; off >>= 1) part += __shfl_down(part, off, 32);
    if (lane == 0) atomicAdd(score, part);
}

// -------------------- per-edge-type node attention logits --------------------
template<int H, int D>
__global__ void k_att_ab(const float* __restrict__ hs, const float* __restrict__ hd,
                         const float* __restrict__ as_, const float* __restrict__ ad_,
                         float* __restrict__ asrc, float* __restrict__ adst)
{
    int i = blockIdx.x * blockDim.x + threadIdx.x;
    if (i >= NN * H) return;
    int n = i / H, h = i % H;
    const float* sp = hs + (size_t)n * (H * D) + h * D;
    const float* dp = hd + (size_t)n * (H * D) + h * D;
    float s = 0.f, d2 = 0.f;
#pragma unroll
    for (int d = 0; d < D; d++) {
        s  += sp[d] * as_[h * D + d];
        d2 += dp[d] * ad_[h * D + d];
    }
    asrc[i] = s;
    adst[i] = d2;
}

// -------------------- edge pass 1: leaky-relu logits + segment max --------------------
template<int H>
__global__ void k_edge_alpha(const int* __restrict__ row, const int* __restrict__ col,
                             const float* __restrict__ asrc, const float* __restrict__ adst,
                             float* __restrict__ alpha, float* __restrict__ amax)
{
    int i = blockIdx.x * blockDim.x + threadIdx.x;
    if (i >= NE * H) return;
    int j = i / H, h = i % H;
    float a = asrc[row[j] * H + h] + adst[col[j] * H + h];
    a = a > 0.f ? a : 0.2f * a;
    alpha[i] = a;
    float* addr = &amax[col[j] * H + h];
    if (a >= 0.f) atomicMax((int*)addr, __float_as_int(a));
    else          atomicMin((unsigned int*)addr, (unsigned int)__float_as_int(a));
}

// -------------------- edge pass 2: exp + segment sum --------------------
template<int H>
__global__ void k_edge_exp(const int* __restrict__ col, const float* __restrict__ amax,
                           float* __restrict__ alpha, float* __restrict__ denom)
{
    int i = blockIdx.x * blockDim.x + threadIdx.x;
    if (i >= NE * H) return;
    int j = i / H, h = i % H;
    float ex = expf(alpha[i] - amax[col[j] * H + h]);
    alpha[i] = ex;
    atomicAdd(&denom[col[j] * H + h], ex);
}

// -------------------- edge pass 3: weighted message scatter-add --------------------
template<int H, int D>
__global__ void k_edge_scatter(const int* __restrict__ row, const int* __restrict__ col,
                               const float* __restrict__ alpha, const float* __restrict__ denom,
                               const float* __restrict__ hs, float* __restrict__ agg)
{
    constexpr int C = H * D, CG = C / 4;
    long long i = (long long)blockIdx.x * blockDim.x + threadIdx.x;
    if (i >= (long long)NE * CG) return;
    int j  = (int)(i / CG);
    int cg = (int)(i % CG) * 4;
    int h  = cg / D;
    int r = row[j], c = col[j];
    float w = alpha[(size_t)j * H + h] / (denom[(size_t)c * H + h] + 1e-16f);
    const float* hp = hs + (size_t)r * C + cg;
    float* ap = agg + (size_t)c * C + cg;
    float4 hv = *(const float4*)hp;
    atomicAdd(ap + 0, hv.x * w);
    atomicAdd(ap + 1, hv.y * w);
    atomicAdd(ap + 2, hv.z * w);
    atomicAdd(ap + 3, hv.w * w);
}

// -------------------- softmax over relations, grouped by dst type --------------------
__global__ void k_rel_softmax(const float* __restrict__ score, float* __restrict__ sattn)
{
    int t = threadIdx.x;
    if (t >= 4) return;
    const int DSTT[10] = {0, 1, 2, 3, 1, 0, 1, 2, 1, 3};
    float mx = -1e30f;
#pragma unroll
    for (int e = 0; e < 10; e++) if (DSTT[e] == t && score[e] > mx) mx = score[e];
    float s = 0.f;
#pragma unroll
    for (int e = 0; e < 10; e++) if (DSTT[e] == t) s += expf(score[e] - mx);
#pragma unroll
    for (int e = 0; e < 10; e++) if (DSTT[e] == t) sattn[e] = expf(score[e] - mx) / s;
}

// -------------------- layer-1 fusion: weighted relu sum, then elu --------------------
template<int C>
__global__ void k_fuse_elu(const float* __restrict__ outs, const float* __restrict__ sattn,
                           float* __restrict__ dst)
{
    int i = blockIdx.x * blockDim.x + threadIdx.x;
    if (i >= 4 * NN * C) return;
    int t  = i / (NN * C);
    int rc = i % (NN * C);
    const int DSTT[10] = {0, 1, 2, 3, 1, 0, 1, 2, 1, 3};
    float acc = 0.f;
#pragma unroll
    for (int e = 0; e < 10; e++)
        if (DSTT[e] == t)
            acc += sattn[e] * fmaxf(outs[(size_t)e * NN * C + rc], 0.f);
    dst[i] = acc > 0.f ? acc : (expf(acc) - 1.f);
}

// -------------------- layer-2 fusion + final channel softmax --------------------
__global__ void k_fuse2_softmax(const float* __restrict__ outs, const float* __restrict__ sattn,
                                float* __restrict__ out)
{
    int i = blockIdx.x * blockDim.x + threadIdx.x;   // one thread per (type, node)
    if (i >= 4 * NN) return;
    int t = i / NN, n = i % NN;
    const int DSTT[10] = {0, 1, 2, 3, 1, 0, 1, 2, 1, 3};
    float v[32];
#pragma unroll
    for (int c = 0; c < 32; c++) v[c] = 0.f;
#pragma unroll
    for (int e = 0; e < 10; e++)
        if (DSTT[e] == t) {
            const float* p = outs + (size_t)e * NN * 32 + (size_t)n * 32;
            float se = sattn[e];
            for (int c = 0; c < 32; c++) v[c] += se * fmaxf(p[c], 0.f);
        }
    float mx = -1e30f;
#pragma unroll
    for (int c = 0; c < 32; c++) mx = fmaxf(mx, v[c]);
    float s = 0.f;
#pragma unroll
    for (int c = 0; c < 32; c++) { v[c] = expf(v[c] - mx); s += v[c]; }
    float inv = 1.f / s;
    float* o = out + ((size_t)t * NN + n) * 32;
#pragma unroll
    for (int c = 0; c < 32; c++) o[c] = v[c] * inv;
}

// ==================== host orchestration ====================
static const int H_SRC[10] = {0, 1, 2, 3, 0, 1, 2, 1, 3, 1};
static const int H_DSTh[10] = {0, 1, 2, 3, 1, 0, 1, 2, 1, 3};

extern "C" void kernel_launch(void* const* d_in, const int* in_sizes, int n_in,
                              void* d_out, int out_size, void* d_ws, size_t ws_size,
                              hipStream_t stream)
{
    const float* x[4] = {(const float*)d_in[0], (const float*)d_in[1],
                         (const float*)d_in[2], (const float*)d_in[3]};
    const int*   edges    = (const int*)d_in[4];
    const float* proj1_W  = (const float*)d_in[5];
    const float* proj1_b  = (const float*)d_in[6];
    const float* att1_src = (const float*)d_in[7];
    const float* att1_dst = (const float*)d_in[8];
    const float* k1_W     = (const float*)d_in[9];
    const float* k1_b     = (const float*)d_in[10];
    const float* q1       = (const float*)d_in[11];
    const float* proj2_W  = (const float*)d_in[12];
    const float* proj2_b  = (const float*)d_in[13];
    const float* att2_src = (const float*)d_in[14];
    const float* att2_dst = (const float*)d_in[15];
    const float* k2_W     = (const float*)d_in[16];
    const float* k2_b     = (const float*)d_in[17];
    const float* q2       = (const float*)d_in[18];

    // ---- workspace carve-up (floats; every region start is 32B aligned) ----
    const size_t NC1 = (size_t)NN * 128, NC2 = (size_t)NN * 32;
    float* ws     = (float*)d_ws;
    float* h1     = ws;                 // [4][NN][128]  (reused as layer-2 input)
    float* outs1  = h1 + 4 * NC1;       // [10][NN][128]
    float* h2     = outs1 + 10 * NC1;   // [4][NN][32]
    float* outs2  = h2 + 4 * NC2;       // [10][NN][32]
    float* asrc   = outs2 + 10 * NC2;   // [NN*8]
    float* adst   = asrc + NN * 8;
    float* amax   = adst + NN * 8;
    float* denom  = amax + NN * 8;
    float* alpha  = denom + NN * 8;     // [NE*8]
    float* scores = alpha + (size_t)NE * 8;  // [16]
    float* sattn  = scores + 16;             // [16]
    __bf16* p1Wp  = (__bf16*)(sattn + 16);   // packed weights (bf16, WMMA-B order)
    __bf16* k1Wp  = p1Wp + 4 * 256 * 128;
    __bf16* p2Wp  = k1Wp + 128 * 128;
    __bf16* k2Wp  = p2Wp + 4 * 128 * 32;

    const int TB = 256;
    const float invN = 1.0f / (float)NN;
    dim3 w32(32);

    // ---- pre-swizzle all weights into WMMA B-fragment order (tiny, L2-resident) ----
    k_pack_w<<<(4 * 256 * 128 + TB - 1) / TB, TB, 0, stream>>>(proj1_W, p1Wp, 256, 128, 4);
    k_pack_w<<<(128 * 128     + TB - 1) / TB, TB, 0, stream>>>(k1_W,    k1Wp, 128, 128, 1);
    k_pack_w<<<(4 * 128 * 32  + TB - 1) / TB, TB, 0, stream>>>(proj2_W, p2Wp, 128, 32,  4);
    k_pack_w<<<(32 * 32       + TB - 1) / TB, TB, 0, stream>>>(k2_W,    k2Wp, 32,  32,  1);

    // ================= Layer 1 (H=8, D=16, C=128, K=256) =================
    k_proj_wmma<256, 128><<<dim3(NN / 16, 4), w32, 0, stream>>>(
        x[0], x[1], x[2], x[3], p1Wp, proj1_b, h1);

    k_fill<<<(int)((10 * NC1 + TB - 1) / TB), TB, 0, stream>>>(outs1, 0.f, (int)(10 * NC1));

    for (int e = 0; e < 10; e++) {
        const int s = H_SRC[e], d = H_DSTh[e];
        const int* row = edges + (size_t)e * 2 * NE;
        const int* col = row + NE;

        k_att_ab<8, 16><<<(NN * 8 + TB - 1) / TB, TB, 0, stream>>>(
            h1 + (size_t)s * NC1, h1 + (size_t)d * NC1,
            att1_src + e * 128, att1_dst + e * 128, asrc, adst);
        k_fill<<<(NN * 8 + TB - 1) / TB, TB, 0, stream>>>(amax, -INFINITY, NN * 8);
        k_fill<<<(NN * 8 + TB - 1) / TB, TB, 0, stream>>>(denom, 0.f, NN * 8);
        k_edge_alpha<8><<<(NE * 8 + TB - 1) / TB, TB, 0, stream>>>(row, col, asrc, adst, alpha, amax);
        k_edge_exp<8><<<(NE * 8 + TB - 1) / TB, TB, 0, stream>>>(col, amax, alpha, denom);
        k_edge_scatter<8, 16><<<(int)(((long long)NE * 32 + TB - 1) / TB), TB, 0, stream>>>(
            row, col, alpha, denom, h1 + (size_t)s * NC1, outs1 + (size_t)e * NC1);
    }

    k_fill<<<1, 32, 0, stream>>>(scores, 0.f, 16);
    for (int e = 0; e < 10; e++)
        k_sem_score_wmma<128><<<NN / 16, w32, 0, stream>>>(
            outs1 + (size_t)e * NC1, k1Wp, k1_b, q1, scores + e, invN);
    k_rel_softmax<<<1, 4, 0, stream>>>(scores, sattn);
    k_fuse_elu<128><<<(int)((4 * NC1 + TB - 1) / TB), TB, 0, stream>>>(outs1, sattn, h1);

    // ================= Layer 2 (H=1, D=32, C=32, K=128) =================
    k_proj_wmma<128, 32><<<dim3(NN / 16, 4), w32, 0, stream>>>(
        h1, h1 + NC1, h1 + 2 * NC1, h1 + 3 * NC1, p2Wp, proj2_b, h2);

    k_fill<<<(int)((10 * NC2 + TB - 1) / TB), TB, 0, stream>>>(outs2, 0.f, (int)(10 * NC2));

    for (int e = 0; e < 10; e++) {
        const int s = H_SRC[e], d = H_DSTh[e];
        const int* row = edges + (size_t)e * 2 * NE;
        const int* col = row + NE;

        k_att_ab<1, 32><<<(NN + TB - 1) / TB, TB, 0, stream>>>(
            h2 + (size_t)s * NC2, h2 + (size_t)d * NC2,
            att2_src + e * 32, att2_dst + e * 32, asrc, adst);
        k_fill<<<(NN + TB - 1) / TB, TB, 0, stream>>>(amax, -INFINITY, NN);
        k_fill<<<(NN + TB - 1) / TB, TB, 0, stream>>>(denom, 0.f, NN);
        k_edge_alpha<1><<<(NE + TB - 1) / TB, TB, 0, stream>>>(row, col, asrc, adst, alpha, amax);
        k_edge_exp<1><<<(NE + TB - 1) / TB, TB, 0, stream>>>(col, amax, alpha, denom);
        k_edge_scatter<1, 32><<<(int)(((long long)NE * 8 + TB - 1) / TB), TB, 0, stream>>>(
            row, col, alpha, denom, h2 + (size_t)s * NC2, outs2 + (size_t)e * NC2);
    }

    k_fill<<<1, 32, 0, stream>>>(scores, 0.f, 16);
    for (int e = 0; e < 10; e++)
        k_sem_score_wmma<32><<<NN / 16, w32, 0, stream>>>(
            outs2 + (size_t)e * NC2, k2Wp, k2_b, q2, scores + e, invN);
    k_rel_softmax<<<1, 4, 0, stream>>>(scores, sattn);

    k_fuse2_softmax<<<(4 * NN + TB - 1) / TB, TB, 0, stream>>>(outs2, sattn, (float*)d_out);
}
